// AugmentedODE_79233556677072
// MI455X (gfx1250) — matrix operand order repaired and checked by reading, not compiled, not run on hardware
//
#include <hip/hip_runtime.h>

// ---------------------------------------------------------------------------
// AugmentedODE on MI455X (gfx1250): 5 fused batched GEMMs, f16 WMMA, f32 accum
//   K1:  ST  = 0.5*(A^T - A) - G^T u     (== S^T; makes K2's B-stage contiguous)
//   K3:  M2  = lam G^T                   (f16 intermediate, L2-resident)
//   K2:  du  = G + u S                   (B-tile via GLOBAL_LOAD_ASYNC_TO_LDS)
//   K4:  dlam= lam A + (M2 + M2^T) u     (two K-loops, symmetrization in staging)
// Software-pipelined, double-buffered LDS, one barrier per k-step. All staging
// uses single-base-address + immediate-offset load/store clauses. Outputs use
// non-temporal stores so f16 intermediates (67MB) stay resident in 192MB L2.
// ---------------------------------------------------------------------------

typedef __attribute__((ext_vector_type(16))) _Float16 v16h;
typedef __attribute__((ext_vector_type(8)))  _Float16 v8h;
typedef __attribute__((ext_vector_type(4)))  _Float16 v4h;
typedef __attribute__((ext_vector_type(8)))  float    v8f;

#define DDIM 512
#define NBATCH 64
#define LDA 40          // LDS row stride in halfs: 80 B, multiple of 16 B

constexpr int MODE_M1ST = 0;   // ST = 0.5(A^T - A) - G^T u
constexpr int MODE_DU   = 1;   // du = G + u S        (S[k][n] = ST[n][k])
constexpr int MODE_M2   = 2;   // M2 = lam G^T
constexpr int MODE_DLAM = 3;   // dlam = lam A + (M2 + M2^T) u

union F16Frag { v16h v; v8h h[2]; };

union TileReg {
  float4   d[4];   // direct f32 path
  float    s[16];  // transpose f32 path
  _Float16 y[16];  // symmetrized f16 path
};

__device__ __forceinline__ void wait_asynccnt0() {
#if __has_builtin(__builtin_amdgcn_s_wait_asynccnt)
  __builtin_amdgcn_s_wait_asynccnt(0);
#else
  asm volatile("s_wait_asynccnt 0" ::: "memory");
#endif
}

// ---- A-operand fragment, 16x32 f16, ISA register layout ----
__device__ __forceinline__ v16h load_afrag(const _Float16* lds, int m0, int lane) {
  const int m  = lane & 15;
  const int kb = (lane >> 4) * 8;
  const _Float16* base = lds + (m0 + m) * LDA;
  F16Frag f;
  f.h[0] = *(const v8h*)(base + kb);
  f.h[1] = *(const v8h*)(base + 16 + kb);
  return f.v;
}

// ---- B-operand fragment, 32x16 f16; LDS holds B^T (N-major, K contiguous) ----
__device__ __forceinline__ v16h load_bfrag(const _Float16* lds, int n0, int lane) {
  const int n  = lane & 15;
  const int kb = (lane >> 4) * 8;
  const _Float16* base = lds + (n0 + n) * LDA + kb;
  F16Frag f;
  f.h[0] = *(const v8h*)(base);
  f.h[1] = *(const v8h*)(base + 16);
  return f.v;
}

// ============ register staging: one base address + immediate offsets ============
__device__ __forceinline__ void ld_direct_f32(TileReg& r, const float* src, int tid) {
  const float* p = src + (size_t)(tid >> 3) * DDIM + (tid & 7) * 4;
#pragma unroll
  for (int c = 0; c < 4; ++c)
    r.d[c] = *(const float4*)(p + (size_t)c * (32 * DDIM));
}
__device__ __forceinline__ void st_direct_f32(_Float16* dst, const TileReg& r, int tid) {
  _Float16* q = dst + (tid >> 3) * LDA + (tid & 7) * 4;
#pragma unroll
  for (int c = 0; c < 4; ++c) {
    const float4 f = r.d[c];
    v4h h = { (_Float16)f.x, (_Float16)f.y, (_Float16)f.z, (_Float16)f.w };
    *(v4h*)(q + c * (32 * LDA)) = h;
  }
}
__device__ __forceinline__ void ld_trans_f32(TileReg& r, const float* src, int tid) {
  const float* p = src + (size_t)(tid >> 7) * DDIM + (tid & 127);
#pragma unroll
  for (int c = 0; c < 16; ++c)
    r.s[c] = p[(size_t)c * (2 * DDIM)];
}
__device__ __forceinline__ void st_trans_f32(_Float16* dst, const TileReg& r, int tid) {
  _Float16* q = dst + (tid & 127) * LDA + (tid >> 7);
#pragma unroll
  for (int c = 0; c < 16; ++c)
    q[c * 2] = (_Float16)r.s[c];
}
__device__ __forceinline__ void ld_sym_f16(TileReg& r, const _Float16* m2b,
                                           int cm0, int k0, int tid) {
  const _Float16* pd = m2b + (size_t)(cm0 + (tid >> 5)) * DDIM + k0 + (tid & 31);
  const _Float16* pt = m2b + (size_t)(k0 + (tid & 31)) * DDIM + cm0 + (tid >> 5);
#pragma unroll
  for (int c = 0; c < 16; ++c)
    r.y[c] = (_Float16)((float)pd[(size_t)c * (8 * DDIM)] + (float)pt[c * 8]);
}
__device__ __forceinline__ void st_16h(_Float16* dst, const TileReg& r, int tid) {
  _Float16* q = dst + (tid >> 5) * LDA + (tid & 31);
#pragma unroll
  for (int c = 0; c < 16; ++c)
    q[c * (8 * LDA)] = r.y[c];
}

// ---- CDNA5 async global->LDS copy for the contiguous f16 tile (ASYNCcnt) ----
// Each thread copies two 16B chunks of a 128x32-half tile. The ISA adds the
// instruction IOFFSET to BOTH addresses, so each chunk gets its own registers.
__device__ __forceinline__ void async_ld_f16_tile(_Float16* dstLds,
                                                  const _Float16* src, int tid) {
  _Float16* l0 = dstLds + (tid >> 2) * LDA + (tid & 3) * 8;
  const _Float16* g0 = src + (size_t)(tid >> 2) * DDIM + (tid & 3) * 8;
  const unsigned lo0 = (unsigned)(uintptr_t)l0;           // low 32b = LDS offset
  const unsigned lo1 = lo0 + 64 * LDA * 2;                // +64 rows
  const _Float16* g1 = g0 + (size_t)64 * DDIM;
  asm volatile("global_load_async_to_lds_b128 %0, %1, off"
               :: "v"(lo0), "v"(g0) : "memory");
  asm volatile("global_load_async_to_lds_b128 %0, %1, off"
               :: "v"(lo1), "v"(g1) : "memory");
}

template <int MODE>
__global__ __launch_bounds__(256) void gemm_kernel(
    const float* __restrict__ U, const float* __restrict__ LAM,
    const float* __restrict__ AM, const float* __restrict__ GM,
    _Float16* __restrict__ STB, _Float16* __restrict__ M2B,
    float* __restrict__ DUO, float* __restrict__ DLO) {
  __shared__ _Float16 lA[2][128 * LDA];
  __shared__ _Float16 lB[2][128 * LDA];

  const int tid  = threadIdx.x;
  const int lane = tid & 31;
  const int wave = tid >> 5;
  const int wm   = (wave & 1) * 64;
  const int wn   = (wave >> 1) * 32;
  const int cm0  = blockIdx.y * 128;
  const int cn0  = blockIdx.x * 128;
  const size_t off = (size_t)blockIdx.z * DDIM * DDIM;

  const float* u_b  = U   + off;
  const float* l_b  = LAM + off;
  const float* a_b  = AM  + off;
  const float* g_b  = GM  + off;
  _Float16*    st_b = STB + off;
  _Float16*    m2_b = M2B + off;

  // Issue all global traffic for tile t (B-tile of DU goes async straight to LDS).
  auto load_tiles = [&](int t, TileReg& ra, TileReg& rb, _Float16* ldsB) {
    if constexpr (MODE == MODE_M1ST) {
      const int k0 = t * 32;
      ld_trans_f32(ra, g_b + (size_t)k0 * DDIM + cm0, tid);      // A-op = G^T
      ld_trans_f32(rb, u_b + (size_t)k0 * DDIM + cn0, tid);      // B-op = u
    } else if constexpr (MODE == MODE_DU) {
      const int k0 = t * 32;
      ld_direct_f32(ra, u_b + (size_t)cm0 * DDIM + k0, tid);     // A-op = u
      async_ld_f16_tile(ldsB, st_b + (size_t)cn0 * DDIM + k0, tid); // B^T = ST
    } else if constexpr (MODE == MODE_M2) {
      const int k0 = t * 32;
      ld_direct_f32(ra, l_b + (size_t)cm0 * DDIM + k0, tid);     // A-op = lam
      ld_direct_f32(rb, g_b + (size_t)cn0 * DDIM + k0, tid);     // B^T = G rows
    } else {
      const int ph = t >> 4, k0 = (t & 15) * 32;
      if (ph == 0) {
        ld_direct_f32(ra, l_b + (size_t)cm0 * DDIM + k0, tid);   // A-op = lam
        ld_trans_f32(rb, a_b + (size_t)k0 * DDIM + cn0, tid);    // B-op = A
      } else {
        ld_sym_f16(ra, m2_b, cm0, k0, tid);                      // A-op = M2+M2^T
        ld_trans_f32(rb, u_b + (size_t)k0 * DDIM + cn0, tid);    // B-op = u
      }
    }
  };

  auto store_tiles = [&](int t, _Float16* dA, _Float16* dB, TileReg& ra, TileReg& rb) {
    if constexpr (MODE == MODE_M1ST) {
      st_trans_f32(dA, ra, tid); st_trans_f32(dB, rb, tid);
    } else if constexpr (MODE == MODE_DU) {
      st_direct_f32(dA, ra, tid);                 // B-tile already in flight (async)
    } else if constexpr (MODE == MODE_M2) {
      st_direct_f32(dA, ra, tid); st_direct_f32(dB, rb, tid);
    } else {
      if ((t >> 4) == 0) { st_direct_f32(dA, ra, tid); st_trans_f32(dB, rb, tid); }
      else               { st_16h(dA, ra, tid);        st_trans_f32(dB, rb, tid); }
    }
  };

  v8f acc[4][2] = {};
  const int T = (MODE == MODE_DLAM) ? 32 : 16;

  TileReg ra, rb;
  load_tiles(0, ra, rb, lB[0]);
  store_tiles(0, lA[0], lB[0], ra, rb);
  if constexpr (MODE == MODE_DU) wait_asynccnt0();
  __syncthreads();

  for (int t = 0; t < T; ++t) {
    const int cur = t & 1, nxt = cur ^ 1;
    if (t + 1 < T) load_tiles(t + 1, ra, rb, lB[nxt]);  // overlap with WMMA below

    v16h af[4];
    v16h bf[2];
#pragma unroll
    for (int i = 0; i < 4; ++i) af[i] = load_afrag(lA[cur], wm + 16 * i, lane);
#pragma unroll
    for (int j = 0; j < 2; ++j) bf[j] = load_bfrag(lB[cur], wn + 16 * j, lane);
#pragma unroll
    for (int i = 0; i < 4; ++i)
#pragma unroll
      for (int j = 0; j < 2; ++j)
        acc[i][j] = __builtin_amdgcn_wmma_f32_16x16x32_f16(
            false, af[i], false, bf[j], (short)0, acc[i][j], false, false);

    if (t + 1 < T) store_tiles(t + 1, lA[nxt], lB[nxt], ra, rb);
    if constexpr (MODE == MODE_DU) wait_asynccnt0();
    __syncthreads();
  }

  // ---- epilogue: one base address per stream, immediate offsets ----
  const int hb = (lane >> 4) * 8;
  const int nc = lane & 15;
  const size_t obase = (size_t)(cm0 + wm + hb) * DDIM + (cn0 + wn + nc);  // [r][c]
  const size_t tbase = (size_t)(cn0 + wn + nc) * DDIM + (cm0 + wm + hb);  // [c][r]
#pragma unroll
  for (int i = 0; i < 4; ++i)
#pragma unroll
    for (int j = 0; j < 2; ++j)
#pragma unroll
      for (int v = 0; v < 8; ++v) {
        const size_t o  = obase + (size_t)(16 * i + v) * DDIM + 16 * j;
        const float val = acc[i][j][v];
        if constexpr (MODE == MODE_M1ST) {
          const size_t ot = tbase + (size_t)(16 * j) * DDIM + 16 * i + v;
          st_b[o] = (_Float16)(0.5f * (a_b[ot] - a_b[o]) - val);
        } else if constexpr (MODE == MODE_DU) {
          __builtin_nontemporal_store(g_b[o] + val, &DUO[off + o]);
        } else if constexpr (MODE == MODE_M2) {
          m2_b[o] = (_Float16)val;
        } else {
          __builtin_nontemporal_store(val, &DLO[off + o]);
        }
      }
}

extern "C" void kernel_launch(void* const* d_in, const int* in_sizes, int n_in,
                              void* d_out, int out_size, void* d_ws, size_t ws_size,
                              hipStream_t stream) {
  (void)in_sizes; (void)n_in; (void)out_size; (void)ws_size;
  const float* u   = (const float*)d_in[0];
  const float* lam = (const float*)d_in[1];
  const float* A   = (const float*)d_in[2];
  const float* G   = (const float*)d_in[3];
  // d_in[4] = t (unused by the reference math)

  const size_t per = (size_t)NBATCH * DDIM * DDIM;
  float* du   = (float*)d_out;
  float* dlam = du + per;

  _Float16* ST = (_Float16*)d_ws;        // 33.5 MB, stays in L2
  _Float16* M2 = ST + per;               // 33.5 MB   (67 MB total workspace)

  dim3 grid(DDIM / 128, DDIM / 128, NBATCH);
  dim3 block(256);

  gemm_kernel<MODE_M1ST><<<grid, block, 0, stream>>>(u, lam, A, G, ST, M2, du, dlam);
  gemm_kernel<MODE_M2  ><<<grid, block, 0, stream>>>(u, lam, A, G, ST, M2, du, dlam);
  gemm_kernel<MODE_DU  ><<<grid, block, 0, stream>>>(u, lam, A, G, ST, M2, du, dlam);
  gemm_kernel<MODE_DLAM><<<grid, block, 0, stream>>>(u, lam, A, G, ST, M2, du, dlam);
}
